// AnchorRefine_128849018984
// MI455X (gfx1250) — matrix operand order repaired and checked by reading, not compiled, not run on hardware
//
#include <hip/hip_runtime.h>
#include <hip/hip_bf16.h>
#include <math.h>

// ---------------- problem constants (from reference) ----------------
#define K_ANCH   22500
#define N_IMG    32
#define M_GT     50
#define NM       (N_IMG * M_GT)       // 1600
#define IMG_SZ   800.0f
#define POS_TH   0.7f
#define NEG_TH   0.3f
#define MAX_FG   128
#define TOTAL_A  256
#define POS_CAP  8192                 // per-row positive list capacity (ws + LDS bound)

typedef _Float16     v16h __attribute__((ext_vector_type(16)));
typedef float        v8f  __attribute__((ext_vector_type(8)));
typedef unsigned int v4u  __attribute__((ext_vector_type(4)));
typedef int          v8i  __attribute__((ext_vector_type(8)));
typedef int          v4i  __attribute__((ext_vector_type(4)));

// Deterministic priority hash (surrogate for jax.random.uniform; monotone key).
__device__ __forceinline__ unsigned keyhash(unsigned n, unsigned k, unsigned salt) {
  unsigned x = k * 0x9E3779B9u + n * 0x85EBCA6Bu + salt * 0xC2B2AE35u + 0x01234567u;
  x ^= x >> 16; x *= 0x7FEB352Du;
  x ^= x >> 15; x *= 0x846CA68Bu;
  x ^= x >> 16;
  return x;
}

// Classic +1-pixel IoU; identical code in both passes -> bit-identical values.
__device__ __forceinline__ float iou_pp(float4 a, float4 g) {
  float areaA = (a.z - a.x + 1.f) * (a.w - a.y + 1.f);
  float areaG = (g.z - g.x + 1.f) * (g.w - g.y + 1.f);
  float iw = fminf(a.z, g.z) - fmaxf(a.x, g.x) + 1.f;
  float ih = fminf(a.w, g.w) - fmaxf(a.y, g.y) + 1.f;
  iw = fmaxf(iw, 0.f); ih = fmaxf(ih, 0.f);
  float inter = iw * ih;
  return inter / (areaA + areaG - inter);
}

// ---------------- K1: per-(n,m) max IoU over all anchors; zero posCnt ----------------
__global__ void k1_gmax(const float4* __restrict__ anchors,
                        const float4* __restrict__ gt,
                        float* __restrict__ gmax,
                        int* __restrict__ posCnt) {
  __shared__ float sW[8];
  int tid = threadIdx.x;
  if (blockIdx.x == 0 && tid < N_IMG) posCnt[tid] = 0;   // reset scratch each call
  float4 g = gt[blockIdx.x];
  float best = -1.f;
  for (int k = tid; k < K_ANCH; k += 256) {
    if (k + 256 < K_ANCH) __builtin_prefetch(&anchors[k + 256], 0, 1); // global_prefetch_b8
    best = fmaxf(best, iou_pp(anchors[k], g));
  }
  for (int off = 16; off; off >>= 1) best = fmaxf(best, __shfl_xor(best, off, 32));
  if ((tid & 31) == 0) sW[tid >> 5] = best;
  __syncthreads();
  if (tid == 0) {
    float m = sW[0];
    for (int w = 1; w < 8; w++) m = fmaxf(m, sW[w]);
    gmax[blockIdx.x] = m;
  }
}

// ---------------- K2: labels + regression coeffs; TDM stages gt boxes into LDS ----------------
struct __align__(16) K2Smem {
  float gt[NM * 4];   // MUST be first: TDM destination at LDS offset 0
  float gmax[NM];
};

__global__ void k2_assign(const float4* __restrict__ anchors,
                          const float*  __restrict__ gtRaw,
                          const float*  __restrict__ gmax,
                          float* __restrict__ labels,
                          float4* __restrict__ coeff,
                          int* __restrict__ posCnt,
                          int* __restrict__ posIdx) {
  __shared__ K2Smem sm;
  int tid = threadIdx.x;

#if defined(__HIP_DEVICE_COMPILE__) && defined(__gfx1250__) && \
    __has_builtin(__builtin_amdgcn_tensor_load_to_lds) && \
    __has_builtin(__builtin_amdgcn_s_wait_tensorcnt)
  {
    // Tensor DMA descriptor (D#): 2-D tensor, data_size=4B, dim0=tile0=6400 elems,
    // dim1=tile1=1, LDS dest offset 0. Every wave issues (TDM ignores EXEC; per-wave
    // TENSORcnt means every wave must wait on its own copy anyway).
    unsigned long long ga = (unsigned long long)(const void*)gtRaw;
    const unsigned TD0 = NM * 4;                       // 6400 elements
    v4u g0;
    g0[0] = 1u;                                        // count=1, user mode, no gather
    g0[1] = 0u;                                        // lds_addr = 0
    g0[2] = (unsigned)(ga & 0xFFFFFFFFu);              // global_addr[31:0]
    g0[3] = (unsigned)((ga >> 32) & 0x01FFFFFFu) | (2u << 30);  // addr[56:32] | type=2
    v8i g1;
    g1[0] = (int)(2u << 16);                           // wg_mask=0, data_size=4B
    g1[1] = (int)((TD0 & 0xFFFFu) << 16);              // tensor_dim0[15:0]
    g1[2] = (int)(((TD0 >> 16) & 0xFFFFu) | (1u << 16)); // dim0[31:16] | tensor_dim1=1
    g1[3] = (int)((TD0 & 0xFFFFu) << 16);              // dim1[31:16]=0 | tile_dim0
    g1[4] = 1;                                         // tile_dim1=1, tile_dim2=0
    g1[5] = (int)TD0;                                  // tensor_dim0_stride[31:0]
    g1[6] = 0;                                         // stride0[47:32] | stride1[15:0]
    g1[7] = 0;
    v4i gz4 = {0, 0, 0, 0};
    v8i gz8 = {0, 0, 0, 0, 0, 0, 0, 0};                // unused trailing group (zeros)
    __builtin_amdgcn_tensor_load_to_lds(g0, g1, gz4, gz4, gz8, 0);
    __builtin_amdgcn_s_wait_tensorcnt(0);
  }
#else
  for (int i = tid; i < NM * 4; i += 256) sm.gt[i] = gtRaw[i];
#endif
  for (int i = tid; i < NM; i += 256) sm.gmax[i] = gmax[i];
  __syncthreads();

  int k = blockIdx.x * 256 + tid;
  if (k >= K_ANCH) return;
  float4 a = anchors[k];
  const float4* gtv = (const float4*)sm.gt;

  for (int n = 0; n < N_IMG; n++) {
    float best = -1.f; int bestM = 0;
    bool aBox = false, anyPos = false, anyNeg = false;
    #pragma unroll 5
    for (int m = 0; m < M_GT; m++) {
      float v = iou_pp(a, gtv[n * M_GT + m]);
      if (v > best) { best = v; bestM = m; }   // first-max, matches jnp.argmax
      aBox   |= (v == sm.gmax[n * M_GT + m]);
      anyPos |= (v >= POS_TH);
      anyNeg |= (v >= NEG_TH);
    }
    float lab = -1.f;
    if (!anyNeg) lab = 0.f;
    if (aBox || anyPos) lab = 1.f;
    labels[n * K_ANCH + k] = lab;
    if (lab == 1.f) {
      int slot = atomicAdd(&posCnt[n], 1);
      if (slot < POS_CAP) posIdx[n * POS_CAP + slot] = k;
    }
    float4 g = gtv[n * M_GT + bestM];
    float aw = a.z - a.x + 1.f, ah = a.w - a.y + 1.f;
    float ax = a.x + 0.5f * aw, ay = a.y + 0.5f * ah;
    float gw = g.z - g.x + 1.f, gh = g.w - g.y + 1.f;
    float gx = g.x + 0.5f * gw, gy = g.y + 0.5f * gh;
    float4 r;
    r.x = (gx - ax) / aw; r.y = (gy - ay) / ah;
    r.z = logf(gw / aw);  r.w = logf(gh / ah);
    coeff[n * K_ANCH + k] = r;
  }
}

// ---------------- K3: fg subsample — ranks via comparison-matrix x ones using WMMA ----------------
// rank_i = sum_j [key_j < key_i].  A = 16x32 f16 tile of 0/1 comparisons (ISA A-layout),
// B = ones, f32 accumulator carries exact counts -> v_wmma_f32_16x16x32_f16.
__global__ void k3_fg(float* __restrict__ labels,
                      const int* __restrict__ posCnt,
                      const int* __restrict__ posIdx) {
  __shared__ unsigned sKey[POS_CAP];
  int n = blockIdx.x;
  int tid = threadIdx.x;                      // 128 threads = 4 waves
  int P = posCnt[n]; if (P > POS_CAP) P = POS_CAP;
  if (P <= MAX_FG) return;                    // all positives kept
  int Pr = (P + 31) & ~31;
  for (int j = tid; j < Pr; j += 128)
    sKey[j] = (j < P) ? keyhash((unsigned)n, (unsigned)posIdx[n * POS_CAP + j], 1u)
                      : 0xFFFFFFFFu;          // pad: never strictly-less
  __syncthreads();

  int lane = tid & 31, wave = tid >> 5;
  int row  = lane & 15;                       // M-row this lane owns
  int hi   = lane >> 4;                       // lane group (K-range select)

  for (int ti = wave; ti * 16 < P; ti += 4) { // uniform per wave -> EXEC all ones
    int i = ti * 16 + row;
    unsigned myKey = (i < P) ? sKey[i] : 0u;
    v8f acc = {0.f, 0.f, 0.f, 0.f, 0.f, 0.f, 0.f, 0.f};
    v16h ones;
    #pragma unroll
    for (int t = 0; t < 16; t++) ones[t] = (_Float16)1.0f;

    for (int j0 = 0; j0 < Pr; j0 += 32) {
      v16h av;
      #pragma unroll
      for (int t = 0; t < 16; t++) {
        // 16-bit A-matrix 16x32 layout: lanes0-15: K = t (t<8) else t+8
        //                               lanes16-31: K = t+8 (t<8) else t+16
        int kk = (t < 8) ? (t + (hi ? 8 : 0)) : (t + 8 + (hi ? 8 : 0));
        av[t] = (sKey[j0 + kk] < myKey) ? (_Float16)1.0f : (_Float16)0.0f;
      }
      acc = __builtin_amdgcn_wmma_f32_16x16x32_f16(
          /*neg_a=*/false, av, /*neg_b=*/false, ones,
          /*c_mod=*/(short)0, acc, /*reuse_a=*/false, /*reuse_b=*/false);
    }
    // D layout: lanes0-15 hold M=r in acc[r]; lanes16-31 hold M=8+r.
    if (lane == 0 || lane == 16) {
      int base = ti * 16 + hi * 8;
      #pragma unroll
      for (int r = 0; r < 8; r++) {
        int i2 = base + r;
        if (i2 < P && acc[r] >= (float)MAX_FG)
          labels[n * K_ANCH + posIdx[n * POS_CAP + i2]] = -1.f;  // rank >= 128 -> drop
      }
    }
  }
}

// ---------------- K4: bg subsample — 32-step radix select of maxBg smallest keys ----------------
__global__ void k4_bg(float* __restrict__ labels, const int* __restrict__ posCnt) {
  __shared__ int sP[9];
  int n = blockIdx.x, tid = threadIdx.x;
  int lane = tid & 31, wave = tid >> 5;
  int P = posCnt[n];
  int numFg = P < MAX_FG ? P : MAX_FG;
  int maxBg = TOTAL_A - numFg;

  int cnt = 0;
  for (int k = tid; k < K_ANCH; k += 256) cnt += (labels[n * K_ANCH + k] == 0.f);
  for (int off = 16; off; off >>= 1) cnt += __shfl_xor(cnt, off, 32);
  if (lane == 0) sP[wave] = cnt;
  __syncthreads();
  if (tid == 0) { int t = 0; for (int w = 0; w < 8; w++) t += sP[w]; sP[8] = t; }
  __syncthreads();
  int bgTotal = sP[8];
  if (bgTotal <= maxBg) return;               // keep all bg
  __syncthreads();

  // Find key X with rank (maxBg-1); keep bg keys <= X (keys effectively distinct).
  unsigned prefix = 0;
  for (int b = 31; b >= 0; b--) {
    unsigned cand = prefix | (1u << b);
    int c = 0;
    for (int k = tid; k < K_ANCH; k += 256) {
      if (labels[n * K_ANCH + k] == 0.f &&
          keyhash((unsigned)n, (unsigned)k, 2u) < cand) c++;
    }
    for (int off = 16; off; off >>= 1) c += __shfl_xor(c, off, 32);
    if (lane == 0) sP[wave] = c;
    __syncthreads();
    if (tid == 0) { int t = 0; for (int w = 0; w < 8; w++) t += sP[w]; sP[8] = t; }
    __syncthreads();
    if (sP[8] <= maxBg - 1) prefix = cand;
    __syncthreads();
  }
  for (int k = tid; k < K_ANCH; k += 256) {
    if (labels[n * K_ANCH + k] == 0.f &&
        keyhash((unsigned)n, (unsigned)k, 2u) > prefix)
      labels[n * K_ANCH + k] = -1.f;
  }
}

// ---------------- K5: anchors_idx = stable compaction of in-image anchors ----------------
__global__ void k5_idx(const float4* __restrict__ anchors, int* __restrict__ outIdx) {
  __shared__ int sBase;
  __shared__ int sWc[8];
  int tid = threadIdx.x, lane = tid & 31, wave = tid >> 5;
  for (int k = tid; k < K_ANCH; k += 256) outIdx[k] = -1;
  if (tid == 0) sBase = 0;
  __syncthreads();
  for (int c0 = 0; c0 < K_ANCH; c0 += 256) {
    int k = c0 + tid;
    bool keep = false;
    if (k < K_ANCH) {
      float4 a = anchors[k];
      keep = (a.x >= 0.f) && (a.y >= 0.f) && (a.z < IMG_SZ) && (a.w < IMG_SZ);
    }
    unsigned m32 = (unsigned)__ballot(keep);  // wave32: low 32 bits
    int laneOff = __popc(m32 & ((1u << lane) - 1u));
    if (lane == 0) sWc[wave] = __popc(m32);
    __syncthreads();
    int wOff = 0;
    for (int w = 0; w < wave; w++) wOff += sWc[w];
    if (keep) outIdx[sBase + wOff + laneOff] = k;
    __syncthreads();
    if (tid == 0) { int t = 0; for (int w = 0; w < 8; w++) t += sWc[w]; sBase += t; }
    __syncthreads();
  }
}

// ---------------- launcher ----------------
extern "C" void kernel_launch(void* const* d_in, const int* in_sizes, int n_in,
                              void* d_out, int out_size, void* d_ws, size_t ws_size,
                              hipStream_t stream) {
  const float4* anchors = (const float4*)d_in[0];
  const float*  gtRaw   = (const float*)d_in[1];
  const float4* gtv4    = (const float4*)d_in[1];

  float*  labels = (float*)d_out;
  float4* coeff  = (float4*)((float*)d_out + (size_t)N_IMG * K_ANCH);
  int*    outIdx = (int*)((float*)d_out + (size_t)N_IMG * K_ANCH * 5);

  // ws: gmax[1600] f32 | posCnt[32] i32 | posIdx[32*8192] i32  (~1.06 MB)
  float* gmax   = (float*)d_ws;
  int*   posCnt = (int*)d_ws + NM;
  int*   posIdx = posCnt + N_IMG;

  k1_gmax  <<<NM, 256, 0, stream>>>(anchors, gtv4, gmax, posCnt);
  k2_assign<<<(K_ANCH + 255) / 256, 256, 0, stream>>>(anchors, gtRaw, gmax,
                                                      labels, coeff, posCnt, posIdx);
  k3_fg    <<<N_IMG, 128, 0, stream>>>(labels, posCnt, posIdx);
  k4_bg    <<<N_IMG, 256, 0, stream>>>(labels, posCnt);
  k5_idx   <<<1, 256, 0, stream>>>(anchors, outIdx);
}
